// RGCN_LSTM_VAE_51599737094225
// MI455X (gfx1250) — matrix operand order repaired and checked by reading, not compiled, not run on hardware
//
#include <hip/hip_runtime.h>
#include <hip/hip_bf16.h>

typedef __attribute__((ext_vector_type(16))) _Float16 v16h;
typedef __attribute__((ext_vector_type(8)))  _Float16 v8h;
typedef __attribute__((ext_vector_type(2)))  _Float16 h2;
typedef __attribute__((ext_vector_type(8)))  float    v8f;

#define T_DIM   8192
#define N_NODES 10
#define FIN     7
#define FOUT    64
#define H_DIM   512
#define LATD    128
#define E_NUM   40
#define R_NUM   2
#define NF      70      // N_NODES*FIN
#define NO      640     // N_NODES*FOUT
#define G4      2048    // 4*H
#define KPAD    96      // NF padded to multiple of 32
#define NPAD2   80      // fc2 out padded to multiple of 16
#define LBLK    16      // persistent LSTM blocks
#define LTHR    256

// ---------------------------------------------------------------- helpers
union HU { unsigned short u; _Float16 f; };

__device__ inline v8f wmma_tile_f16(const _Float16* Abase, int lda,
                                    const _Float16* Bbase, int ldb, int K) {
  const int lane = threadIdx.x & 31;
  const int hsel = lane >> 4;     // 0: lanes 0-15, 1: lanes 16-31
  const int lr   = lane & 15;
  v8f c = {};
  for (int k0 = 0; k0 < K; k0 += 32) {
    union { v16h v; v8h p[2]; } a, b;
    // A (16xK f16, ISA 7.12.2): lanes0-15 row=lr hold K {0..7,16..23};
    // lanes16-31 hold K {8..15,24..31}  (per 32-wide K slice)
    const _Float16* pa = Abase + lr * lda + k0 + hsel * 8;
    a.p[0] = *(const v8h*)(pa);
    a.p[1] = *(const v8h*)(pa + 16);
    // B (Kx16 f16), stored column(N)-major with K contiguous:
    // lanes0-15 col=lr hold K {0..7,16..23}; lanes16-31 hold K {8..15,24..31}
    const _Float16* pb = Bbase + lr * ldb + k0 + hsel * 8;
    b.p[0] = *(const v8h*)(pb);
    b.p[1] = *(const v8h*)(pb + 16);
    c = __builtin_amdgcn_wmma_f32_16x16x32_f16(false, a.v, false, b.v,
                                               (short)0, c, false, false);
  }
  return c;
}

// ---------------------------------------------------------------- phase 0: fold RGCN into M (70x640)
__global__ void build_M(const float* __restrict__ W_rel, const float* __restrict__ W_root,
                        const int* __restrict__ ei, const int* __restrict__ et,
                        float* __restrict__ Mc) {
  int idx = blockIdx.x * blockDim.x + threadIdx.x;
  if (idx >= NF * NO) return;
  int k = idx / NO, j = idx % NO;
  int nsrc = k / FIN, f = k % FIN;
  int ndst = j / FOUT, o = j % FOUT;
  float cnt[R_NUM] = {0.f, 0.f};
  for (int e = 0; e < E_NUM; ++e)
    if (ei[E_NUM + e] == ndst) cnt[et[e]] += 1.f;
  float acc = (nsrc == ndst) ? W_root[f * FOUT + o] : 0.f;
  for (int e = 0; e < E_NUM; ++e) {
    if (ei[e] == nsrc && ei[E_NUM + e] == ndst) {
      int r = et[e];
      acc += W_rel[(r * FIN + f) * FOUT + o] / fmaxf(cnt[r], 1.f);
    }
  }
  Mc[idx] = acc;
}

// ---------------------------------------------------------------- phase 0b: W_comb = M @ W_ih.T  (f16, n-major, K padded)
__global__ void fuse_Wcomb(const float* __restrict__ Mc, const float* __restrict__ W_ih,
                           const float* __restrict__ b_rgcn, const float* __restrict__ b_ih,
                           const float* __restrict__ b_hh,
                           _Float16* __restrict__ Wc, float* __restrict__ biasc) {
  int idx = blockIdx.x * blockDim.x + threadIdx.x;
  if (idx >= G4 * KPAD) return;
  int n = idx / KPAD, k = idx % KPAD;
  if (k < NF) {
    float acc = 0.f;
    for (int j = 0; j < NO; ++j) acc += Mc[k * NO + j] * W_ih[n * NO + j];
    Wc[idx] = (_Float16)acc;
  } else {
    Wc[idx] = (_Float16)0.f;
    if (k == NF) {  // one thread per n computes fused bias (b_ih + b_hh + rgcn bias thru W_ih)
      float acc = b_ih[n] + b_hh[n];
      for (int j = 0; j < NO; ++j) acc += b_rgcn[j & 63] * W_ih[n * NO + j];
      biasc[n] = acc;
    }
  }
}

// ---------------------------------------------------------------- converts
__global__ void cvt_f16(const float* __restrict__ s, _Float16* __restrict__ d, int n) {
  int i = blockIdx.x * blockDim.x + threadIdx.x;
  if (i < n) d[i] = (_Float16)s[i];
}
__global__ void cvt_x(const float* __restrict__ x, _Float16* __restrict__ d) {
  int i = blockIdx.x * blockDim.x + threadIdx.x;
  if (i >= T_DIM * KPAD) return;
  int t = i / KPAD, k = i % KPAD;
  d[i] = (k < NF) ? (_Float16)x[t * NF + k] : (_Float16)0.f;
}
__global__ void cvt_fc2(const float* __restrict__ s, _Float16* __restrict__ d) {
  int i = blockIdx.x * blockDim.x + threadIdx.x;
  if (i >= NPAD2 * LATD) return;
  int n = i / LATD, k = i % LATD;
  d[i] = (n < NF) ? (_Float16)s[n * LATD + k] : (_Float16)0.f;
}
__global__ void init_bar(int* bar) { *bar = 0; }

// ---------------------------------------------------------------- GEMM1: gates_x = Xf16 @ Wcomb + biasc  (8192 x 2048, K=96)
__global__ void gemm_gates(const _Float16* __restrict__ X, const _Float16* __restrict__ W,
                           const float* __restrict__ biasc, float* __restrict__ out) {
  int mt = blockIdx.x, nt = blockIdx.y;
  v8f c = wmma_tile_f16(X + mt * 16 * KPAD, KPAD, W + nt * 16 * KPAD, KPAD, KPAD);
  int lane = threadIdx.x & 31, hsel = lane >> 4, lr = lane & 15;
  int n = nt * 16 + lr;
  float bb = biasc[n];
#pragma unroll
  for (int v = 0; v < 8; ++v) {
    int m = mt * 16 + hsel * 8 + v;   // C/D layout: VGPR v -> M=v (lanes<16) / v+8
    out[m * G4 + n] = c[v] + bb;
  }
}

// ---------------------------------------------------------------- LSTM recurrence: 16 persistent blocks, W_hh slice LDS-resident
__global__ void __launch_bounds__(LTHR, 1)
lstm_persist(const _Float16* __restrict__ Whh, const float* __restrict__ gx,
             _Float16* hs, int* bar) {
  __shared__ _Float16 Wl[128 * H_DIM];   // 128 KB: this block's 128 gate rows
  __shared__ _Float16 hl[H_DIM];
  __shared__ float psum[LTHR];
  __shared__ float gl[128];
  const int b = blockIdx.x, tid = threadIdx.x;
  // stage W_hh slice: local row lr -> global gate row (lr/32)*512 + b*32 + lr%32
  for (int i = tid; i < 128 * H_DIM; i += LTHR) {
    int lr = i >> 9, k = i & (H_DIM - 1);
    int gr = (lr >> 5) * H_DIM + b * 32 + (lr & 31);
    Wl[i] = Whh[gr * H_DIM + k];
  }
  float cst = 0.f;
  // warm the first gates_x row into cache while W stages
  if (tid < 128) {
    int gr0 = (tid >> 5) * H_DIM + b * 32 + (tid & 31);
    __builtin_prefetch(&gx[gr0], 0, 1);   // -> global_prefetch_b8
  }
  __syncthreads();
  for (int t = 0; t < T_DIM; ++t) {
    // acquire h(t-1): hs row t (written by all blocks at step t-1)
    if (t > 0) {
      if (tid == 0) {
        while (__hip_atomic_load(bar, __ATOMIC_ACQUIRE, __HIP_MEMORY_SCOPE_AGENT) < LBLK * t)
          __builtin_amdgcn_s_sleep(1);
      }
      __syncthreads();
      for (int i = tid; i < H_DIM; i += LTHR) {
        HU cv; cv.u = ((const volatile unsigned short*)hs)[t * H_DIM + i];
        hl[i] = cv.f;
      }
    } else {
      for (int i = tid; i < H_DIM; i += LTHR) hl[i] = (_Float16)0.f;
    }
    __syncthreads();
    // dot: 128 rows x 512, two threads per row (f16 pair dot -> f32 accumulate)
    {
      int lr = tid >> 1, kh = tid & 1;
      const _Float16* wr = &Wl[lr * H_DIM + kh * 256];
      const _Float16* hr = &hl[kh * 256];
      float acc = 0.f;
#if __has_builtin(__builtin_amdgcn_fdot2)
      const h2* wr2 = (const h2*)wr;
      const h2* hr2 = (const h2*)hr;
#pragma unroll 8
      for (int k = 0; k < 128; ++k)
        acc = __builtin_amdgcn_fdot2(wr2[k], hr2[k], acc, false);
#else
#pragma unroll 8
      for (int k = 0; k < 256; ++k) acc += (float)wr[k] * (float)hr[k];
#endif
      psum[tid] = acc;
    }
    __syncthreads();
    if (tid < 128) {
      int gr = (tid >> 5) * H_DIM + b * 32 + (tid & 31);
      gl[tid] = psum[2 * tid] + psum[2 * tid + 1] + gx[(size_t)t * G4 + gr];
      if (t + 1 < T_DIM)                       // hide next row's HBM/L2 latency
        __builtin_prefetch(&gx[(size_t)(t + 1) * G4 + gr], 0, 1);
    }
    __syncthreads();
    if (tid < 32) {
      float gi = gl[tid], gf = gl[32 + tid], gg = gl[64 + tid], go = gl[96 + tid];
      float si = 1.f / (1.f + __expf(-gi));
      float sf = 1.f / (1.f + __expf(-gf));
      float so = 1.f / (1.f + __expf(-go));
      cst = sf * cst + si * tanhf(gg);
      float h = so * tanhf(cst);
      hs[(t + 1) * H_DIM + b * 32 + tid] = (_Float16)h;
      __threadfence();
    }
    __syncthreads();
    if (tid == 0)
      __hip_atomic_fetch_add(bar, 1, __ATOMIC_RELEASE, __HIP_MEMORY_SCOPE_AGENT);
    __syncthreads();
  }
}

// ---------------------------------------------------------------- GEMM2: hfc = relu(hs @ fc1_w.T + b) -> mu | log_var
__global__ void gemm_fc1(const _Float16* __restrict__ hs1, const _Float16* __restrict__ W,
                         const float* __restrict__ b1, float* __restrict__ muP,
                         float* __restrict__ lvP) {
  int mt = blockIdx.x, nt = blockIdx.y;
  v8f c = wmma_tile_f16(hs1 + mt * 16 * H_DIM, H_DIM, W + nt * 16 * H_DIM, H_DIM, H_DIM);
  int lane = threadIdx.x & 31, hsel = lane >> 4, lr = lane & 15;
  int n = nt * 16 + lr;
  float bb = b1[n];
#pragma unroll
  for (int v = 0; v < 8; ++v) {
    int m = mt * 16 + hsel * 8 + v;
    float val = fmaxf(c[v] + bb, 0.f);
    if (n < LATD) muP[m * LATD + n] = val;
    else          lvP[m * LATD + (n - LATD)] = val;
  }
}

// ---------------------------------------------------------------- z = mu + eps * exp(0.5*log_var)
__global__ void z_kernel(const float* __restrict__ muP, const float* __restrict__ lvP,
                         const float* __restrict__ eps, _Float16* __restrict__ z) {
  int i = blockIdx.x * blockDim.x + threadIdx.x;
  if (i >= T_DIM * LATD) return;
  z[i] = (_Float16)(muP[i] + eps[i] * __expf(0.5f * lvP[i]));
}

// ---------------------------------------------------------------- GEMM3: rec = sigmoid(z @ fc2_w.T + b)
__global__ void gemm_fc2(const _Float16* __restrict__ Z, const _Float16* __restrict__ W,
                         const float* __restrict__ b2, float* __restrict__ rec) {
  int mt = blockIdx.x, nt = blockIdx.y;
  v8f c = wmma_tile_f16(Z + mt * 16 * LATD, LATD, W + nt * 16 * LATD, LATD, LATD);
  int lane = threadIdx.x & 31, hsel = lane >> 4, lr = lane & 15;
  int n = nt * 16 + lr;
  if (n >= NF) return;
  float bb = b2[n];
#pragma unroll
  for (int v = 0; v < 8; ++v) {
    int m = mt * 16 + hsel * 8 + v;
    float val = c[v] + bb;
    rec[m * NF + n] = 1.f / (1.f + __expf(-val));
  }
}

// ================================================================ host
extern "C" void kernel_launch(void* const* d_in, const int* in_sizes, int n_in,
                              void* d_out, int out_size, void* d_ws, size_t ws_size,
                              hipStream_t stream) {
  const float* x      = (const float*)d_in[0];
  const float* eps    = (const float*)d_in[1];
  const float* W_rel  = (const float*)d_in[2];
  const float* W_root = (const float*)d_in[3];
  const float* b_rgcn = (const float*)d_in[4];
  const float* W_ih   = (const float*)d_in[5];
  const float* W_hh   = (const float*)d_in[6];
  const float* b_ih   = (const float*)d_in[7];
  const float* b_hh   = (const float*)d_in[8];
  const float* fc1_w  = (const float*)d_in[9];
  const float* fc1_b  = (const float*)d_in[10];
  const float* fc2_w  = (const float*)d_in[11];
  const float* fc2_b  = (const float*)d_in[12];
  const int*   ei     = (const int*)d_in[13];
  const int*   et     = (const int*)d_in[14];
  float* out = (float*)d_out;

  // workspace carve-up (256B aligned)
  char* base = (char*)d_ws;
  size_t off = 0;
  auto carve = [&](size_t bytes) { char* p = base + off; off = (off + bytes + 255) & ~(size_t)255; return p; };
  float*    Mc     = (float*)   carve(NF * NO * 4);
  _Float16* Wc16   = (_Float16*)carve((size_t)G4 * KPAD * 2);
  float*    biasc  = (float*)   carve(G4 * 4);
  _Float16* X16    = (_Float16*)carve((size_t)T_DIM * KPAD * 2);
  float*    gx     = (float*)   carve((size_t)T_DIM * G4 * 4);
  _Float16* Whh16  = (_Float16*)carve((size_t)G4 * H_DIM * 2);
  _Float16* hs16   = (_Float16*)carve((size_t)(T_DIM + 1) * H_DIM * 2);  // row 0 = h(-1) slot (unused)
  _Float16* fc1w16 = (_Float16*)carve((size_t)2 * LATD * H_DIM * 2);
  _Float16* fc2w16 = (_Float16*)carve((size_t)NPAD2 * LATD * 2);
  _Float16* z16    = (_Float16*)carve((size_t)T_DIM * LATD * 2);
  int*      bar    = (int*)     carve(256);

  // output regions: rec | mu | log_var
  float* recP = out;
  float* muP  = out + (size_t)T_DIM * NF;
  float* lvP  = out + (size_t)T_DIM * NF + (size_t)T_DIM * LATD;

  // phase 0: fold graph + weights
  build_M<<<(NF * NO + 255) / 256, 256, 0, stream>>>(W_rel, W_root, ei, et, Mc);
  fuse_Wcomb<<<(G4 * KPAD + 255) / 256, 256, 0, stream>>>(Mc, W_ih, b_rgcn, b_ih, b_hh, Wc16, biasc);
  cvt_f16<<<(G4 * H_DIM + 255) / 256, 256, 0, stream>>>(W_hh, Whh16, G4 * H_DIM);
  cvt_f16<<<(2 * LATD * H_DIM + 255) / 256, 256, 0, stream>>>(fc1_w, fc1w16, 2 * LATD * H_DIM);
  cvt_fc2<<<(NPAD2 * LATD + 255) / 256, 256, 0, stream>>>(fc2_w, fc2w16);
  cvt_x<<<(T_DIM * KPAD + 255) / 256, 256, 0, stream>>>(x, X16);
  init_bar<<<1, 1, 0, stream>>>(bar);

  // phase 1: big WMMA GEMM for all input-side gate pre-activations
  gemm_gates<<<dim3(T_DIM / 16, G4 / 16), 32, 0, stream>>>(X16, Wc16, biasc, gx);

  // phase 2: persistent-block LSTM recurrence (LDS-resident W_hh slices)
  lstm_persist<<<LBLK, LTHR, 0, stream>>>(Whh16, gx, hs16, bar);

  // phase 3: VAE head (WMMA)
  gemm_fc1<<<dim3(T_DIM / 16, (2 * LATD) / 16), 32, 0, stream>>>(hs16 + H_DIM, fc1w16, fc1_b, muP, lvP);
  z_kernel<<<(T_DIM * LATD + 255) / 256, 256, 0, stream>>>(muP, lvP, eps, z16);
  gemm_fc2<<<dim3(T_DIM / 16, NPAD2 / 16), 32, 0, stream>>>(z16, fc2w16, fc2_b, recP);
}
